// MultiHeadAttention_72859825209410
// MI455X (gfx1250) — matrix operand order repaired and checked by reading, not compiled, not run on hardware
//
#include <hip/hip_runtime.h>

// ---- problem constants (match reference) ----
constexpr int Bc  = 4;
constexpr int Ec  = 1024;
constexpr int Sc  = 1024;
constexpr int Hc  = 16;
constexpr int HDc = 64;

typedef __attribute__((ext_vector_type(16))) __bf16 v16bf;
typedef __attribute__((ext_vector_type(8)))  __bf16 v8bf;
typedef __attribute__((ext_vector_type(8)))  float  v8f;

#if defined(__has_builtin)
#  if __has_builtin(__builtin_amdgcn_tensor_load_to_lds) && __has_builtin(__builtin_amdgcn_s_wait_tensorcnt)
#    define USE_TDM 1
#  endif
#endif
#ifndef USE_TDM
#  define USE_TDM 0
#endif

#if USE_TDM
typedef __attribute__((ext_vector_type(4))) unsigned int v4u;
typedef __attribute__((ext_vector_type(8))) int          v8i;
typedef __attribute__((ext_vector_type(4))) int          v4i;

// Issue a 2D TDM tile load (bf16 elements) per CDNA5 D# layout
// (CDNA5 ISA). Higher-dim groups zeroed (2D).
// clang-23 toolchain: 6-arg builtin (v4u, v8i, v4i, v4i, v8i, i32 cpol).
__device__ __forceinline__ void tdm_load_tile_2d(
    unsigned lds_byte, const __bf16* gptr,
    unsigned tile_d0, unsigned tile_d1,
    unsigned tens_d0, unsigned tens_d1, unsigned stride0)
{
    unsigned long long ga = (unsigned long long)(size_t)gptr;
    // descriptor words must be wave-uniform scalars
    unsigned galo = (unsigned)__builtin_amdgcn_readfirstlane((int)(unsigned)(ga & 0xFFFFFFFFull));
    unsigned gahi = (unsigned)__builtin_amdgcn_readfirstlane((int)(unsigned)(ga >> 32));
    unsigned lb   = (unsigned)__builtin_amdgcn_readfirstlane((int)lds_byte);

    v4u g0;
    g0.x = 1u;                                      // count=1 (valid user descriptor)
    g0.y = lb;                                      // lds_addr (bytes)
    g0.z = galo;                                    // global_addr[31:0]
    g0.w = (gahi & 0x01FFFFFFu) | (2u << 30);       // global_addr[56:32] | type=2
    v8i g1;
    g1[0] = (int)(1u << 16);                        // wg_mask=0, data_size=1 (2 bytes)
    g1[1] = (int)((tens_d0 & 0xFFFFu) << 16);       // [31:16] = tensor_dim0[15:0]
    g1[2] = (int)(((tens_d0 >> 16) & 0xFFFFu) | ((tens_d1 & 0xFFFFu) << 16));
    g1[3] = (int)(((tens_d1 >> 16) & 0xFFFFu) | ((tile_d0 & 0xFFFFu) << 16));
    g1[4] = (int)(tile_d1 & 0xFFFFu);               // tile_dim1; tile_dim2=0
    g1[5] = (int)stride0;                           // tensor_dim0_stride[31:0]
    g1[6] = 0;                                      // stride0[47:32]=0, stride1 lo=0
    g1[7] = 0;                                      // stride1 hi=0
    v4i z4 = {0, 0, 0, 0};
    v8i z8 = {0, 0, 0, 0, 0, 0, 0, 0};
    __builtin_amdgcn_tensor_load_to_lds(g0, g1, z4, z4, z8, 0);
}
#endif

// ------------------------------------------------------------------
// Kernel 1: f32 -> bf16 conversion (used for Wf, Wq, Wk, Wv)
// ------------------------------------------------------------------
__global__ __launch_bounds__(256) void cvt_kernel(const float* __restrict__ src,
                                                  __bf16* __restrict__ dst, int n) {
    int i = blockIdx.x * 256 + threadIdx.x;
    if (i < n) dst[i] = (__bf16)src[i];
}

// ------------------------------------------------------------------
// Kernel 2: per-head QKV projection, WMMA-based.
// Outputs (bf16):
//   qt[bh][s][d], kt[bh][s][d]  (d-contiguous: A-frag / B-frag of QK^T)
//   vt[bh][d][s]                (s-contiguous: B-frag of PV)
// ------------------------------------------------------------------
__global__ __launch_bounds__(128) void proj_kernel(
    const float* __restrict__ Q, const float* __restrict__ K, const float* __restrict__ V,
    const __bf16* __restrict__ wq, const __bf16* __restrict__ wk, const __bf16* __restrict__ wv,
    __bf16* __restrict__ qt, __bf16* __restrict__ kt, __bf16* __restrict__ vt)
{
    __shared__ __align__(32) __bf16 Xl[64 * 72];   // [s][e], padded row stride 72
    const int tid  = threadIdx.x;
    const int lane = tid & 31;
    const int wvid = tid >> 5;
    const int m    = lane & 15;
    const int half = lane >> 4;
    const int bh = blockIdx.y;
    const int b  = bh >> 4;
    const int h  = bh & 15;
    const int s0 = blockIdx.x * 64;

    for (int which = 0; which < 3; ++which) {
        const float*  X  = (which == 0) ? Q  : (which == 1) ? K  : V;
        const __bf16* Wb = (which == 0) ? wq : (which == 1) ? wk : wv;
        __syncthreads();
        for (int i = tid; i < 64 * 64; i += 128) {
            int e = i >> 6, s = i & 63;
            Xl[s * 72 + e] = (__bf16)X[((size_t)b * Ec + h * HDc + e) * Sc + s0 + s];
        }
        __syncthreads();

        const __bf16* xr = Xl + (size_t)(wvid * 16 + m) * 72;
        v16bf a0, a1;
        {
            v8bf lo0 = *(const v8bf*)(xr + half * 16);
            v8bf hi0 = *(const v8bf*)(xr + half * 16 + 8);
            v8bf lo1 = *(const v8bf*)(xr + 32 + half * 16);
            v8bf hi1 = *(const v8bf*)(xr + 32 + half * 16 + 8);
            #pragma unroll
            for (int i = 0; i < 8; ++i) {
                a0[i] = lo0[i]; a0[i + 8] = hi0[i];
                a1[i] = lo1[i]; a1[i + 8] = hi1[i];
            }
        }

        #pragma unroll
        for (int dsub = 0; dsub < 4; ++dsub) {
            const __bf16* wr = Wb + (size_t)(h * HDc + dsub * 16 + m) * HDc;
            v16bf b0 = *(const v16bf*)(wr + half * 16);
            v16bf b1 = *(const v16bf*)(wr + 32 + half * 16);
            v8f acc = {};
            acc = __builtin_amdgcn_wmma_f32_16x16x32_bf16(false, a0, false, b0, (short)0, acc, false, false);
            acc = __builtin_amdgcn_wmma_f32_16x16x32_bf16(false, a1, false, b1, (short)0, acc, false, false);
            #pragma unroll
            for (int r = 0; r < 8; ++r) {
                const int s = s0 + wvid * 16 + r + 8 * half;
                const int d = dsub * 16 + m;
                __bf16 val = (__bf16)acc[r];
                if (which == 0)      qt[((size_t)bh * Sc + s) * HDc + d] = val;
                else if (which == 1) kt[((size_t)bh * Sc + s) * HDc + d] = val;
                else                 vt[((size_t)bh * HDc + d) * Sc + s] = val;
            }
        }
    }
}

// ------------------------------------------------------------------
// Kernel 3: causal attention, one wave per 16-row query tile.
// K/V tiles staged to LDS via the Tensor Data Mover (double-buffered,
// TENSORcnt-pipelined) when available; B-frags then come from ds_load.
// ------------------------------------------------------------------
__global__ __launch_bounds__(128) void attn_kernel(
    const __bf16* __restrict__ qt, const __bf16* __restrict__ kt,
    const __bf16* __restrict__ vt, __bf16* __restrict__ pl)
{
    __shared__ __align__(32) __bf16 elds[4][16 * 32];  // 1KB per wave
#if USE_TDM
    __shared__ __align__(32) __bf16 kst[4][2][32 * HDc];  // 4KB per wave per buf
    __shared__ __align__(32) __bf16 vst[4][2][HDc * 32];  // 4KB per wave per buf
#endif
    const int lane = threadIdx.x & 31;
    const int wv   = threadIdx.x >> 5;
    const int bh   = blockIdx.y;
    const int ts   = blockIdx.x * 4 + wv;   // 16-row tile index, 0..63
    const int s0   = ts * 16;
    const int m    = lane & 15;
    const int half = lane >> 4;

    const __bf16* kbase = kt + (size_t)bh * Sc * HDc;
    const __bf16* vbase = vt + (size_t)bh * HDc * Sc;

    // A-fragments of q^T (M=s rows, K=d). K chunks: [0,32) and [32,64).
    const __bf16* qbase = qt + ((size_t)bh * Sc + s0) * HDc;
    v16bf aq0, aq1;
    {
        v8bf lo0 = *(const v8bf*)(qbase + m * HDc + half * 16);
        v8bf hi0 = *(const v8bf*)(qbase + m * HDc + half * 16 + 8);
        v8bf lo1 = *(const v8bf*)(qbase + m * HDc + 32 + half * 16);
        v8bf hi1 = *(const v8bf*)(qbase + m * HDc + 32 + half * 16 + 8);
        #pragma unroll
        for (int i = 0; i < 8; ++i) {
            aq0[i] = lo0[i]; aq0[i + 8] = hi0[i];
            aq1[i] = lo1[i]; aq1[i + 8] = hi1[i];
        }
    }

    v16bf bones;
    #pragma unroll
    for (int i = 0; i < 16; ++i) bones[i] = (__bf16)1.0f;

    v8f acc_o[4];
    v8f acc_rs = {};
    #pragma unroll
    for (int t = 0; t < 4; ++t) acc_o[t] = v8f{};

    __bf16* eld = elds[wv];
    const int nj = ts / 2 + 1;                 // causal: 32-key tiles needed

#if USE_TDM
    // Issue K/V tile DMA for a given 32-key tile into buffer (jt&1).
    auto issue_tile = [&](int jt) {
        const int bufi = jt & 1;
        const int t0i  = jt * 32;
        // K tile: 32 rows (t), 64 cols (d), row stride 64
        tdm_load_tile_2d((unsigned)(size_t)&kst[wv][bufi][0],
                         kbase + (size_t)t0i * HDc,
                         /*tile_d0=*/HDc, /*tile_d1=*/32,
                         /*tens_d0=*/HDc, /*tens_d1=*/(unsigned)(Sc - t0i),
                         /*stride0=*/HDc);
        // V tile: 64 rows (d), 32 cols (s), row stride Sc
        tdm_load_tile_2d((unsigned)(size_t)&vst[wv][bufi][0],
                         vbase + t0i,
                         /*tile_d0=*/32, /*tile_d1=*/HDc,
                         /*tens_d0=*/(unsigned)(Sc - t0i), /*tens_d1=*/HDc,
                         /*stride0=*/Sc);
    };
    issue_tile(0);
#endif

    for (int jt = 0; jt < nj; ++jt) {
        const int t0 = jt * 32;
#if USE_TDM
        if (jt + 1 < nj) {
            issue_tile(jt + 1);                       // prefetch next tile
            __builtin_amdgcn_s_wait_tensorcnt(2);     // per-wave TDM in-order:
        } else {                                      // current tile complete
            __builtin_amdgcn_s_wait_tensorcnt(0);
        }
        asm volatile("" ::: "memory");
        const __bf16* kbuf = &kst[wv][jt & 1][0];
        const __bf16* vbuf = &vst[wv][jt & 1][0];
#endif
        // ---- scores: 16(s) x 32(t), K=d=64 ----
        v8f c[2];
        #pragma unroll
        for (int nsub = 0; nsub < 2; ++nsub) {
#if USE_TDM
            const __bf16* kr = kbuf + (size_t)(nsub * 16 + m) * HDc;
#else
            const __bf16* kr = kbase + (size_t)(t0 + nsub * 16 + m) * HDc;
#endif
            v16bf b0 = *(const v16bf*)(kr + half * 16);
            v16bf b1 = *(const v16bf*)(kr + 32 + half * 16);
            v8f z = {};
            z = __builtin_amdgcn_wmma_f32_16x16x32_bf16(false, aq0, false, b0, (short)0, z, false, false);
            z = __builtin_amdgcn_wmma_f32_16x16x32_bf16(false, aq1, false, b1, (short)0, z, false, false);
            c[nsub] = z;
        }
        // ---- scale, exp, causal mask -> LDS (D layout -> row-major tile) ----
        #pragma unroll
        for (int nsub = 0; nsub < 2; ++nsub) {
            const int tcol = t0 + nsub * 16 + m;
            #pragma unroll
            for (int r = 0; r < 8; ++r) {
                const int srow = s0 + r + 8 * half;
                float e = (tcol <= srow) ? __expf(c[nsub][r] * 0.125f) : 0.0f;
                eld[(r + 8 * half) * 32 + nsub * 16 + m] = (__bf16)e;
            }
        }
        // ---- reload E as A-fragment (16 x 32, K=j) ----
        v16bf ae;
        {
            v8bf lo = *(const v8bf*)(eld + m * 32 + half * 16);
            v8bf hi = *(const v8bf*)(eld + m * 32 + half * 16 + 8);
            #pragma unroll
            for (int i = 0; i < 8; ++i) { ae[i] = lo[i]; ae[i + 8] = hi[i]; }
        }
        // ---- row sums via ones-matrix WMMA (same D layout as acc_o) ----
        acc_rs = __builtin_amdgcn_wmma_f32_16x16x32_bf16(false, ae, false, bones, (short)0, acc_rs, false, false);
        // ---- PV: out^T[i][d] += E[i][j] * v^T[j][d] ----
        #pragma unroll
        for (int dsub = 0; dsub < 4; ++dsub) {
#if USE_TDM
            v16bf bv = *(const v16bf*)(vbuf + (size_t)(dsub * 16 + m) * 32 + half * 16);
#else
            v16bf bv = *(const v16bf*)(vbase + (size_t)(dsub * 16 + m) * Sc + t0 + half * 16);
#endif
            acc_o[dsub] = __builtin_amdgcn_wmma_f32_16x16x32_bf16(false, ae, false, bv, (short)0, acc_o[dsub], false, false);
        }
    }

    // ---- normalize and store transposed: pl[b][s][h*64+d] ----
    const int h = bh & 15, b = bh >> 4;
    #pragma unroll
    for (int dsub = 0; dsub < 4; ++dsub)
        #pragma unroll
        for (int r = 0; r < 8; ++r) {
            const int srow = s0 + r + 8 * half;
            const int ecol = h * HDc + dsub * 16 + m;
            pl[((size_t)b * Sc + srow) * Ec + ecol] = (__bf16)(acc_o[dsub][r] / acc_rs[r]);
        }
}

// ------------------------------------------------------------------
// Kernel 4: out[b,e,s] = sum_f Wf[e,f] * pl[b,s,f]
// Wave computes a 64(M=e) x 32(N=s) tile, K step 32 -> 8 WMMAs/step.
// ------------------------------------------------------------------
__global__ __launch_bounds__(128) void fgemm_kernel(
    const __bf16* __restrict__ wf, const __bf16* __restrict__ pl,
    float* __restrict__ out)
{
    const int lane = threadIdx.x & 31;
    const int wv   = threadIdx.x >> 5;
    const int m    = lane & 15;
    const int half = lane >> 4;
    const int b    = blockIdx.z;
    const int m0   = blockIdx.y * 64;
    const int n0   = blockIdx.x * 128 + wv * 32;

    v8f acc[4][2];
    #pragma unroll
    for (int mt = 0; mt < 4; ++mt)
        #pragma unroll
        for (int nt = 0; nt < 2; ++nt) acc[mt][nt] = v8f{};

    const __bf16* plb = pl + (size_t)b * Sc * Ec;

    for (int kk = 0; kk < Ec; kk += 32) {
        if (kk + 32 < Ec) {   // global_prefetch_b8 of next K tile
            __builtin_prefetch((const void*)(wf  + (size_t)(m0 + m) * Ec + kk + 32), 0, 1);
            __builtin_prefetch((const void*)(plb + (size_t)(n0 + m) * Ec + kk + 32), 0, 1);
        }
        v16bf a[4];
        #pragma unroll
        for (int mt = 0; mt < 4; ++mt) {
            const __bf16* ar = wf + (size_t)(m0 + mt * 16 + m) * Ec + kk;
            v8bf lo = *(const v8bf*)(ar + half * 16);
            v8bf hi = *(const v8bf*)(ar + half * 16 + 8);
            #pragma unroll
            for (int i = 0; i < 8; ++i) { a[mt][i] = lo[i]; a[mt][i + 8] = hi[i]; }
        }
        v16bf bb[2];
        #pragma unroll
        for (int nt = 0; nt < 2; ++nt)
            bb[nt] = *(const v16bf*)(plb + (size_t)(n0 + nt * 16 + m) * Ec + kk + half * 16);
        #pragma unroll
        for (int mt = 0; mt < 4; ++mt)
            #pragma unroll
            for (int nt = 0; nt < 2; ++nt)
                acc[mt][nt] = __builtin_amdgcn_wmma_f32_16x16x32_bf16(
                    false, a[mt], false, bb[nt], (short)0, acc[mt][nt], false, false);
    }

    #pragma unroll
    for (int mt = 0; mt < 4; ++mt)
        #pragma unroll
        for (int nt = 0; nt < 2; ++nt)
            #pragma unroll
            for (int r = 0; r < 8; ++r) {
                const int e = m0 + mt * 16 + r + 8 * half;
                const int s = n0 + nt * 16 + m;
                out[((size_t)b * Ec + e) * Sc + s] = acc[mt][nt][r];
            }
}

// ------------------------------------------------------------------
extern "C" void kernel_launch(void* const* d_in, const int* in_sizes, int n_in,
                              void* d_out, int out_size, void* d_ws, size_t ws_size,
                              hipStream_t stream)
{
    const float* Q  = (const float*)d_in[0];
    const float* K  = (const float*)d_in[1];
    const float* V  = (const float*)d_in[2];
    // d_in[3] = mask: causal tril, known statically -> unused
    const float* Wq = (const float*)d_in[4];
    const float* Wk = (const float*)d_in[5];
    const float* Wv = (const float*)d_in[6];
    const float* Wf = (const float*)d_in[7];
    float* out = (float*)d_out;

    const size_t nHead = (size_t)Bc * Hc * Sc * HDc;  // 4,194,304 elems
    const size_t nW    = (size_t)Hc * HDc * HDc;      // 65,536 elems
    __bf16* qt   = (__bf16*)d_ws;
    __bf16* kt   = qt + nHead;
    __bf16* vt   = kt + nHead;
    __bf16* pl   = vt + nHead;                        // [b][s][e]
    __bf16* wfbf = pl + (size_t)Bc * Sc * Ec;         // E*E
    __bf16* wqbf = wfbf + (size_t)Ec * Ec;
    __bf16* wkbf = wqbf + nW;
    __bf16* wvbf = wkbf + nW;

    // 1) weights -> bf16
    cvt_kernel<<<(Ec * Ec + 255) / 256, 256, 0, stream>>>(Wf, wfbf, Ec * Ec);
    cvt_kernel<<<((int)nW + 255) / 256, 256, 0, stream>>>(Wq, wqbf, (int)nW);
    cvt_kernel<<<((int)nW + 255) / 256, 256, 0, stream>>>(Wk, wkbf, (int)nW);
    cvt_kernel<<<((int)nW + 255) / 256, 256, 0, stream>>>(Wv, wvbf, (int)nW);

    // 2) QKV head projections (WMMA) into attention-friendly layouts
    proj_kernel<<<dim3(Sc / 64, Bc * Hc), 128, 0, stream>>>(Q, K, V, wqbf, wkbf, wvbf, qt, kt, vt);

    // 3) causal attention -> pl[b][s][e]
    attn_kernel<<<dim3(Sc / 16 / 4, Bc * Hc), 128, 0, stream>>>(qt, kt, vt, pl);

    // 4) final projection
    fgemm_kernel<<<dim3(Sc / 128, Ec / 64, Bc), 128, 0, stream>>>(wfbf, pl, out);
}
